// MaskedDenseMatMul_50268297232527
// MI455X (gfx1250) — matrix operand order repaired and checked by reading, not compiled, not run on hardware
//
#include <hip/hip_runtime.h>

// MaskedDenseMatMul for MI455X (gfx1250, wave32, WMMA).
//
// qk[bh,q,k] = (sum_d a[bh,q,d]*b[bh,k,d]) * mask[b,q,k]
// B=2,H=8,S=2048,D=64, fp32 in/out.
//
// Roofline: 8.6 GFLOP vs ~300 MB HBM traffic -> memory-bound, ~14us floor
// @ 23.3 TB/s. Matrix work: bf16 16x16x32 WMMA with 3-term split-precision
// (ahi*bhi + alo*bhi + ahi*blo) for ~fp32 accuracy while staying an order
// of magnitude under the memory floor on the matrix pipes. fp32->bf16(hi,lo)
// split happens ONCE per workgroup into LDS (cooperative phase, packed hw
// conversions); waves ds_load_b128 WMMA fragments from LDS (72-element row
// pitch -> conflict-free). Mask loads are batched per N-subtile ahead of the
// WMMA chain so their latency hides under matrix work instead of a
// serialized load->wait->mul->store chain.

#define DEVINL __device__ __forceinline__

typedef __attribute__((ext_vector_type(16))) __bf16 v16bf;
typedef __attribute__((ext_vector_type(8)))  float  v8f;
typedef __attribute__((ext_vector_type(4)))  float  f32x4;
typedef __attribute__((ext_vector_type(4)))  __bf16 bf16x4;

union Frag {
    v16bf v;
    uint4 q[2];
};

constexpr int S    = 2048;
constexpr int D    = 64;
constexpr int TILE = 128;
constexpr int PAD  = 72;   // bf16 elements per LDS row (144B pitch, conflict-free)

// Convert one float4 into (hi, lo) bf16 splits and store 8B each to LDS.
// convertvector lets the backend use packed bf16 converts.
DEVINL void convert_store4(float4 x, unsigned short* hi, unsigned short* lo) {
    f32x4 f = {x.x, x.y, x.z, x.w};
    bf16x4 h4 = __builtin_convertvector(f, bf16x4);            // RNE
    f32x4  hf = __builtin_convertvector(h4, f32x4);
    bf16x4 l4 = __builtin_convertvector(f - hf, bf16x4);
    *(bf16x4*)hi = h4;
    *(bf16x4*)lo = l4;
}

// Fragment load from LDS: two 16B chunks at element offsets k0 and k0+step.
DEVINL void lds_frag(const unsigned short* s, int row, int k0, int step, Frag& f) {
    const unsigned short* base = s + row * PAD + k0;
    f.q[0] = *(const uint4*)(base);
    f.q[1] = *(const uint4*)(base + step);
}

__global__ __launch_bounds__(256)
void masked_mm_wmma_kernel(const float* __restrict__ A,
                           const float* __restrict__ Bm,
                           const float* __restrict__ Mk,
                           float* __restrict__ O) {
    __shared__ __align__(16) unsigned short sAhi[TILE * PAD];
    __shared__ __align__(16) unsigned short sAlo[TILE * PAD];
    __shared__ __align__(16) unsigned short sBhi[TILE * PAD];
    __shared__ __align__(16) unsigned short sBlo[TILE * PAD];

    const int bh  = blockIdx.z;        // fused batch*head, 0..15
    const int bat = bh >> 3;           // batch index (H=8) for mask broadcast
    const int m0  = blockIdx.x * TILE; // query-tile origin
    const int n0  = blockIdx.y * TILE; // key-tile origin

    const float* Abh = A  + (size_t)bh  * S * D;
    const float* Bbh = Bm + (size_t)bh  * S * D;
    const float* Mb  = Mk + (size_t)bat * S * S;
    float*       Obh = O  + (size_t)bh  * S * S;

    // ---- Phase 1: cooperative load + fp32->bf16(hi,lo) split into LDS ----
    // 16 threads per row (16B float4 each -> 256B/row), 16 rows per sweep,
    // 8 sweeps for 128 rows. Coalesced global_load_b128.
    {
        const int c4 = threadIdx.x & 15;   // float4 column (covers 64 floats)
        const int r0 = threadIdx.x >> 4;   // row within 16-row sweep
#pragma unroll
        for (int it = 0; it < 8; ++it) {
            const int row = it * 16 + r0;
            const float4 av = *(const float4*)(Abh + (size_t)(m0 + row) * D + c4 * 4);
            const float4 bv = *(const float4*)(Bbh + (size_t)(n0 + row) * D + c4 * 4);
            convert_store4(av, &sAhi[row * PAD + c4 * 4], &sAlo[row * PAD + c4 * 4]);
            convert_store4(bv, &sBhi[row * PAD + c4 * 4], &sBlo[row * PAD + c4 * 4]);
        }
    }
    __syncthreads();

    // ---- Phase 2: WMMA from LDS fragments --------------------------------
    const int wave = threadIdx.x >> 5;  // 8 waves: 4 (M) x 2 (N)
    const int lane = threadIdx.x & 31;
    const int half = lane >> 4;         // lane half selects K sub-range
    const int r    = lane & 15;         // row/col within 16-wide tile
    const int wm   = wave & 3;          // 32-row strip within 128
    const int wn   = wave >> 2;         // 64-col strip within 128

    // A fragments, resident across all N-subtiles.
    // 16-bit A 16x32 layout: lane<16 holds K 0..7 (v0-3) & 16..23 (v4-7),
    // lane>=16 holds K 8..15 & 24..31 -> chunks at k0=ks*32+half*8, step 16.
    Frag ahi[2][2], alo[2][2];          // [m-subtile][k-step]
#pragma unroll
    for (int mt = 0; mt < 2; ++mt) {
        const int rowA = wm * 32 + mt * 16 + r;
#pragma unroll
        for (int ks = 0; ks < 2; ++ks) {
            lds_frag(sAhi, rowA, ks * 32 + half * 8, 16, ahi[mt][ks]);
            lds_frag(sAlo, rowA, ks * 32 + half * 8, 16, alo[mt][ks]);
        }
    }

#pragma unroll
    for (int nt = 0; nt < 4; ++nt) {
        const int kcol = n0 + wn * 64 + nt * 16;
        const int col  = kcol + r;

        // Batch ALL mask loads for this N-subtile up front (independent
        // registers -> one load clause, latency hidden under the WMMAs).
        float mv[2][8];
#pragma unroll
        for (int mt = 0; mt < 2; ++mt) {
            const int q0t = m0 + wm * 32 + mt * 16 + half * 8;
            const float* mp = Mb + (size_t)q0t * S + col;
#pragma unroll
            for (int i = 0; i < 8; ++i) {
                mv[mt][i] = mp[(size_t)i * S];
            }
        }

        // B (KxN) layout: lane n<16 holds col N=n, K 0..7 & 16..23;
        // lane n+16 holds K 8..15 & 24..31 -> chunks at k0=ks*32+half*8, step 16.
        Frag bhi[2], blo[2];
        const int rowB = wn * 64 + nt * 16 + r;
#pragma unroll
        for (int ks = 0; ks < 2; ++ks) {
            lds_frag(sBhi, rowB, ks * 32 + half * 8, 16, bhi[ks]);
            lds_frag(sBlo, rowB, ks * 32 + half * 8, 16, blo[ks]);
        }

#pragma unroll
        for (int mt = 0; mt < 2; ++mt) {
            v8f c = {};
#pragma unroll
            for (int ks = 0; ks < 2; ++ks) {
                // split-precision: a*b ~= ahi*bhi + alo*bhi + ahi*blo
                c = __builtin_amdgcn_wmma_f32_16x16x32_bf16(
                        false, ahi[mt][ks].v, false, bhi[ks].v,
                        (short)0, c, false, false);
                c = __builtin_amdgcn_wmma_f32_16x16x32_bf16(
                        false, alo[mt][ks].v, false, bhi[ks].v,
                        (short)0, c, false, false);
                c = __builtin_amdgcn_wmma_f32_16x16x32_bf16(
                        false, ahi[mt][ks].v, false, blo[ks].v,
                        (short)0, c, false, false);
            }

            // C/D layout: VGPR i -> row M = half*8 + i, col N = lane&15.
            const int q0t = m0 + wm * 32 + mt * 16 + half * 8;
            float* op = Obh + (size_t)q0t * S + col;
#pragma unroll
            for (int i = 0; i < 8; ++i) {
                op[(size_t)i * S] = c[i] * mv[mt][i];
            }
        }
    }
}

extern "C" void kernel_launch(void* const* d_in, const int* in_sizes, int n_in,
                              void* d_out, int out_size, void* d_ws, size_t ws_size,
                              hipStream_t stream) {
    (void)in_sizes; (void)n_in; (void)out_size; (void)d_ws; (void)ws_size;
    const float* a    = (const float*)d_in[0];
    const float* b    = (const float*)d_in[1];
    const float* mask = (const float*)d_in[2];
    float* out        = (float*)d_out;

    // S/128 = 16 tiles per dim, B*H = 16 batch-heads.
    dim3 grid(16, 16, 16);
    dim3 block(256, 1, 1);  // 8 wave32s
    masked_mm_wmma_kernel<<<grid, block, 0, stream>>>(a, b, mask, out);
}
